// MultiHeadAttention_62551903699412
// MI455X (gfx1250) — matrix operand order repaired and checked by reading, not compiled, not run on hardware
//
#include <hip/hip_runtime.h>

// ---------------------------------------------------------------------------
// MI455X (gfx1250, wave32) multi-head attention:
//   1) QKV projections (fp32 -> f16 WMMA -> f16 head-split, V transposed)
//   2) flash attention (online softmax, f16 WMMA, fp32 accum)
//   3) output projection (f16 WMMA -> fp32 out)
// GEMM kernels use register double-buffering: next K-tile's global loads are
// issued right after the first barrier so they overlap the WMMA block.
// ---------------------------------------------------------------------------

typedef __attribute__((ext_vector_type(16))) _Float16 v16h;
typedef __attribute__((ext_vector_type(8)))  float    v8f;

union AFrag { v16h h; unsigned u[8]; };

__device__ __forceinline__ unsigned pack2h(float a, float b) {
  union { _Float16 h[2]; unsigned u; } t;
  t.h[0] = (_Float16)a; t.h[1] = (_Float16)b;
  return t.u;
}

#define S_LEN 2048
#define DM    1024
#define NH    16
#define DH    64
#define BM    128
#define BN    128
#define BK    32

// ===========================================================================
// Kernel 1: Y = X @ W^T + b for X in {q,k,v}, blockIdx.z selects projection.
//   z=0 -> qh  [b][h][s][64] f16, pre-scaled by 1/sqrt(64)
//   z=1 -> kh  [b][h][s][64] f16
//   z=2 -> vt  [b][h][64][s] f16 (transposed so PV B-frags are contiguous)
// ===========================================================================
__global__ __launch_bounds__(256) void qkv_proj_kernel(
    const float* __restrict__ q,  const float* __restrict__ kk, const float* __restrict__ v,
    const float* __restrict__ wq, const float* __restrict__ wk, const float* __restrict__ wv,
    const float* __restrict__ bq, const float* __restrict__ bk, const float* __restrict__ bv,
    _Float16* __restrict__ qh, _Float16* __restrict__ kh, _Float16* __restrict__ vt)
{
  const int z = blockIdx.z;
  const float* X  = (z == 0) ? q  : ((z == 1) ? kk : v);
  const float* W  = (z == 0) ? wq : ((z == 1) ? wk : wv);
  const float* Bb = (z == 0) ? bq : ((z == 1) ? bk : bv);

  __shared__ _Float16 lA[BM * BK];
  __shared__ _Float16 lB[BN * BK];
  const unsigned* uA = (const unsigned*)lA;
  const unsigned* uB = (const unsigned*)lB;

  const int tid  = threadIdx.x;
  const int wid  = tid >> 5;
  const int lane = tid & 31;
  const int ln   = lane & 15;
  const int hs   = lane >> 4;
  const int wm   = wid & 3;   // 4 waves along M (32 rows each)
  const int wn   = wid >> 2;  // 2 waves along N (64 cols each)
  const int m0   = blockIdx.x * BM;
  const int n0   = blockIdx.y * BN;
  const int row_ld = (tid + 0) >> 3;       // this thread's staged row (p stride 32)
  const int c4_ld  = tid & 7;

  const float4* A4 = (const float4*)X;
  const float4* B4 = (const float4*)W;

  v8f zero = {};
  v8f acc[2][4];
#pragma unroll
  for (int mt = 0; mt < 2; ++mt)
#pragma unroll
    for (int nt = 0; nt < 4; ++nt) acc[mt][nt] = zero;

  // ---- software pipeline: prefetch K-tile k0=0 into registers ----------
  float4 ra[4], rb[4];
#pragma unroll
  for (int p = 0; p < 4; ++p) {
    ra[p] = A4[(size_t)(m0 + row_ld + p * 32) * 256 + c4_ld];
    rb[p] = B4[(size_t)(n0 + row_ld + p * 32) * 256 + c4_ld];
  }

  for (int k0 = 0; k0 < DM; k0 += BK) {
    // Commit prefetched tile to LDS (fp32 -> f16 convert).
#pragma unroll
    for (int p = 0; p < 4; ++p) {
      int idx = tid + p * 256;
      uint2 ta; ta.x = pack2h(ra[p].x, ra[p].y); ta.y = pack2h(ra[p].z, ra[p].w);
      ((uint2*)lA)[idx] = ta;
      uint2 tb; tb.x = pack2h(rb[p].x, rb[p].y); tb.y = pack2h(rb[p].z, rb[p].w);
      ((uint2*)lB)[idx] = tb;
    }
    __syncthreads();

    // Issue next tile's global loads; they overlap the WMMA block below.
    if (k0 + BK < DM) {
      int kn = (k0 + BK) >> 2;
#pragma unroll
      for (int p = 0; p < 4; ++p) {
        ra[p] = A4[(size_t)(m0 + row_ld + p * 32) * 256 + kn + c4_ld];
        rb[p] = B4[(size_t)(n0 + row_ld + p * 32) * 256 + kn + c4_ld];
      }
    }

    AFrag af[2], bf[4];
#pragma unroll
    for (int mt = 0; mt < 2; ++mt) {
      int row = wm * 32 + mt * 16 + ln;
#pragma unroll
      for (int i = 0; i < 8; ++i) {
        int k = ((i & 4) ? 16 : 0) + hs * 8 + 2 * (i & 3);
        af[mt].u[i] = uA[row * 16 + (k >> 1)];
      }
    }
#pragma unroll
    for (int nt = 0; nt < 4; ++nt) {
      int row = wn * 64 + nt * 16 + ln;
#pragma unroll
      for (int i = 0; i < 8; ++i)
        bf[nt].u[i] = uB[row * 16 + ((i & 4) ? 8 : 0) + hs * 4 + (i & 3)];
    }
#pragma unroll
    for (int mt = 0; mt < 2; ++mt)
#pragma unroll
      for (int nt = 0; nt < 4; ++nt)
        acc[mt][nt] = __builtin_amdgcn_wmma_f32_16x16x32_f16(
            false, af[mt].h, false, bf[nt].h, (short)0, acc[mt][nt], false, false);
    __syncthreads();
  }

  // Epilogue: bias + layout-specific f16 store.
#pragma unroll
  for (int mt = 0; mt < 2; ++mt) {
#pragma unroll
    for (int nt = 0; nt < 4; ++nt) {
      int n = n0 + wn * 64 + nt * 16 + ln;
      float bias = Bb[n];
      int h = n >> 6, d = n & 63;
#pragma unroll
      for (int r = 0; r < 8; ++r) {
        int m = m0 + wm * 32 + mt * 16 + r + 8 * hs;
        int b = m >> 11, s = m & (S_LEN - 1);
        float val = acc[mt][nt][r] + bias;
        if (z == 0)
          qh[(((b * NH + h) * S_LEN + s) << 6) + d] = (_Float16)(val * 0.125f);
        else if (z == 1)
          kh[(((b * NH + h) * S_LEN + s) << 6) + d] = (_Float16)val;
        else
          vt[(((b * NH + h) * DH + d) << 11) + s] = (_Float16)val;
      }
    }
  }
}

// ===========================================================================
// Kernel 2: flash attention. Grid (S/128, B*H), 8 waves, 16 queries per wave.
// Online softmax per wave (rows of one wave never cross waves).
// ===========================================================================
__global__ __launch_bounds__(256) void attention_kernel(
    const _Float16* __restrict__ qh, const _Float16* __restrict__ kh,
    const _Float16* __restrict__ vt, _Float16* __restrict__ ctx)
{
  __shared__ _Float16 lP[8 * 16 * 64];   // per-wave P staging (C->A relayout)

  const int tid  = threadIdx.x;
  const int wid  = tid >> 5;
  const int lane = tid & 31;
  const int ln   = lane & 15;
  const int hs   = lane >> 4;
  const int bh   = blockIdx.y;                 // b*16 + h
  const int q0   = blockIdx.x * 128 + wid * 16;

  const unsigned* uQ = (const unsigned*)qh;
  const unsigned* uK = (const unsigned*)kh;
  const unsigned* uV = (const unsigned*)vt;
  const unsigned* uP = (const unsigned*)lP;

  // Q fragments for this wave's 16 rows (k-chunks of 32 over depth 64).
  AFrag qf[2];
#pragma unroll
  for (int c = 0; c < 2; ++c)
#pragma unroll
    for (int i = 0; i < 8; ++i) {
      int d = ((i & 4) ? 16 : 0) + hs * 8 + 2 * (i & 3) + c * 32;
      qf[c].u[i] = uQ[(bh * S_LEN + q0 + ln) * 32 + (d >> 1)];
    }

  v8f zero = {};
  v8f o[4];
#pragma unroll
  for (int t = 0; t < 4; ++t) o[t] = zero;
  float m_run[8], l_run[8];
#pragma unroll
  for (int r = 0; r < 8; ++r) { m_run[r] = -__builtin_inff(); l_run[r] = 0.f; }

  for (int kb = 0; kb < S_LEN; kb += 64) {
    // ---- S = (Q * 1/8) @ K^T -------------------------------------------
    v8f s[4];
#pragma unroll
    for (int t = 0; t < 4; ++t) s[t] = zero;
#pragma unroll
    for (int t = 0; t < 4; ++t) {
#pragma unroll
      for (int c = 0; c < 2; ++c) {
        AFrag kf;
#pragma unroll
        for (int i = 0; i < 8; ++i)
          kf.u[i] = uK[(bh * S_LEN + kb + t * 16 + ln) * 32 + c * 16 + ((i & 4) ? 8 : 0) + hs * 4 + (i & 3)];
        s[t] = __builtin_amdgcn_wmma_f32_16x16x32_f16(
            false, qf[c].h, false, kf.h, (short)0, s[t], false, false);
      }
    }

    // ---- hoist V fragments: global latency hides under softmax VALU ----
    AFrag vf[4][2];
#pragma unroll
    for (int t = 0; t < 4; ++t)
#pragma unroll
      for (int c = 0; c < 2; ++c)
#pragma unroll
        for (int i = 0; i < 8; ++i)
          vf[t][c].u[i] =
              uV[(bh * DH + t * 16 + ln) * 1024 + (kb >> 1) + c * 16 + ((i & 4) ? 8 : 0) + hs * 4 + (i & 3)];

    // ---- online softmax (rows live in 16-lane halves) ------------------
#pragma unroll
    for (int r = 0; r < 8; ++r) {
      float mx = fmaxf(fmaxf(s[0][r], s[1][r]), fmaxf(s[2][r], s[3][r]));
      mx = fmaxf(mx, __shfl_xor(mx, 1));
      mx = fmaxf(mx, __shfl_xor(mx, 2));
      mx = fmaxf(mx, __shfl_xor(mx, 4));
      mx = fmaxf(mx, __shfl_xor(mx, 8));
      float mnew  = fmaxf(m_run[r], mx);
      float alpha = __expf(m_run[r] - mnew);
      m_run[r] = mnew;
      float sum = 0.f;
      int prow = wid * 1024 + (r + 8 * hs) * 64 + ln;
#pragma unroll
      for (int t = 0; t < 4; ++t) {
        float p = __expf(s[t][r] - mnew);
        sum += p;
        lP[prow + t * 16] = (_Float16)p;
        o[t][r] *= alpha;
      }
      sum += __shfl_xor(sum, 1);
      sum += __shfl_xor(sum, 2);
      sum += __shfl_xor(sum, 4);
      sum += __shfl_xor(sum, 8);
      l_run[r] = l_run[r] * alpha + sum;
    }

    // ---- P A-fragments from LDS (same-wave DS ops are in-order) --------
    AFrag pf[2];
#pragma unroll
    for (int c = 0; c < 2; ++c)
#pragma unroll
      for (int i = 0; i < 8; ++i) {
        int j = ((i & 4) ? 16 : 0) + hs * 8 + 2 * (i & 3) + c * 32;
        pf[c].u[i] = uP[wid * 512 + ln * 32 + (j >> 1)];
      }

    // ---- O += P @ V (V stored transposed: contiguous along keys) -------
#pragma unroll
    for (int t = 0; t < 4; ++t)
#pragma unroll
      for (int c = 0; c < 2; ++c)
        o[t] = __builtin_amdgcn_wmma_f32_16x16x32_f16(
            false, pf[c].h, false, vf[t][c].h, (short)0, o[t], false, false);
  }

  // ---- epilogue: normalize, merge heads into ctx [B,S,D] f16 -----------
  const int b = bh >> 4, h = bh & 15;
#pragma unroll
  for (int r = 0; r < 8; ++r) {
    float rinv = 1.f / l_run[r];
    int srow = q0 + r + 8 * hs;
#pragma unroll
    for (int t = 0; t < 4; ++t)
      ctx[(size_t)(b * S_LEN + srow) * DM + h * DH + t * 16 + ln] =
          (_Float16)(o[t][r] * rinv);
  }
}

// ===========================================================================
// Kernel 3: out = ctx @ wo^T + bo  (f16 A, f16 W staged, fp32 out)
// ===========================================================================
__global__ __launch_bounds__(256) void out_proj_kernel(
    const _Float16* __restrict__ ctx, const float* __restrict__ wo,
    const float* __restrict__ bo, float* __restrict__ out)
{
  __shared__ _Float16 lA[BM * BK];
  __shared__ _Float16 lB[BN * BK];
  const unsigned* uA = (const unsigned*)lA;
  const unsigned* uB = (const unsigned*)lB;

  const int tid  = threadIdx.x;
  const int wid  = tid >> 5;
  const int lane = tid & 31;
  const int ln   = lane & 15;
  const int hs   = lane >> 4;
  const int wm   = wid & 3;
  const int wn   = wid >> 2;
  const int m0   = blockIdx.x * BM;
  const int n0   = blockIdx.y * BN;
  const int row_ld = tid >> 3;
  const int c4_ld  = tid & 7;

  const uint2*  A2 = (const uint2*)ctx;
  const float4* B4 = (const float4*)wo;

  v8f zero = {};
  v8f acc[2][4];
#pragma unroll
  for (int mt = 0; mt < 2; ++mt)
#pragma unroll
    for (int nt = 0; nt < 4; ++nt) acc[mt][nt] = zero;

  // ---- software pipeline: prefetch K-tile k0=0 into registers ----------
  uint2  rca[4];
  float4 rb[4];
#pragma unroll
  for (int p = 0; p < 4; ++p) {
    rca[p] = A2[(size_t)(m0 + row_ld + p * 32) * 256 + c4_ld];
    rb[p]  = B4[(size_t)(n0 + row_ld + p * 32) * 256 + c4_ld];
  }

  for (int k0 = 0; k0 < DM; k0 += BK) {
#pragma unroll
    for (int p = 0; p < 4; ++p) {
      int idx = tid + p * 256;
      ((uint2*)lA)[idx] = rca[p];
      uint2 tb; tb.x = pack2h(rb[p].x, rb[p].y); tb.y = pack2h(rb[p].z, rb[p].w);
      ((uint2*)lB)[idx] = tb;
    }
    __syncthreads();

    if (k0 + BK < DM) {
      int kn = (k0 + BK) >> 2;
#pragma unroll
      for (int p = 0; p < 4; ++p) {
        rca[p] = A2[(size_t)(m0 + row_ld + p * 32) * 256 + kn + c4_ld];
        rb[p]  = B4[(size_t)(n0 + row_ld + p * 32) * 256 + kn + c4_ld];
      }
    }

    AFrag af[2], bf[4];
#pragma unroll
    for (int mt = 0; mt < 2; ++mt) {
      int row = wm * 32 + mt * 16 + ln;
#pragma unroll
      for (int i = 0; i < 8; ++i) {
        int k = ((i & 4) ? 16 : 0) + hs * 8 + 2 * (i & 3);
        af[mt].u[i] = uA[row * 16 + (k >> 1)];
      }
    }
#pragma unroll
    for (int nt = 0; nt < 4; ++nt) {
      int row = wn * 64 + nt * 16 + ln;
#pragma unroll
      for (int i = 0; i < 8; ++i)
        bf[nt].u[i] = uB[row * 16 + ((i & 4) ? 8 : 0) + hs * 4 + (i & 3)];
    }
#pragma unroll
    for (int mt = 0; mt < 2; ++mt)
#pragma unroll
      for (int nt = 0; nt < 4; ++nt)
        acc[mt][nt] = __builtin_amdgcn_wmma_f32_16x16x32_f16(
            false, af[mt].h, false, bf[nt].h, (short)0, acc[mt][nt], false, false);
    __syncthreads();
  }

#pragma unroll
  for (int mt = 0; mt < 2; ++mt) {
#pragma unroll
    for (int nt = 0; nt < 4; ++nt) {
      int n = n0 + wn * 64 + nt * 16 + ln;
      float bias = bo[n];
#pragma unroll
      for (int r = 0; r < 8; ++r) {
        int m = m0 + wm * 32 + mt * 16 + r + 8 * hs;
        out[(size_t)m * DM + n] = acc[mt][nt][r] + bias;
      }
    }
  }
}

// ===========================================================================
extern "C" void kernel_launch(void* const* d_in, const int* in_sizes, int n_in,
                              void* d_out, int out_size, void* d_ws, size_t ws_size,
                              hipStream_t stream) {
  (void)in_sizes; (void)n_in; (void)out_size; (void)ws_size;
  const float* q  = (const float*)d_in[0];
  const float* k  = (const float*)d_in[1];
  const float* v  = (const float*)d_in[2];
  const float* wq = (const float*)d_in[3];
  const float* bq = (const float*)d_in[4];
  const float* wk = (const float*)d_in[5];
  const float* bk = (const float*)d_in[6];
  const float* wv = (const float*)d_in[7];
  const float* bv = (const float*)d_in[8];
  const float* wo = (const float*)d_in[9];
  const float* bo = (const float*)d_in[10];
  float* out = (float*)d_out;

  char* ws = (char*)d_ws;
  const size_t HBUF = (size_t)4 * S_LEN * DM * sizeof(_Float16);  // 16 MiB each
  _Float16* qh = (_Float16*)(ws + 0 * HBUF);
  _Float16* kh = (_Float16*)(ws + 1 * HBUF);
  _Float16* vt = (_Float16*)(ws + 2 * HBUF);
  _Float16* cx = (_Float16*)(ws + 3 * HBUF);

  dim3 g1(8192 / BM, DM / BN, 3);     // (64, 8, 3)
  qkv_proj_kernel<<<g1, 256, 0, stream>>>(q, k, v, wq, wk, wv, bq, bk, bv,
                                          qh, kh, vt);
  dim3 g2(S_LEN / 128, 4 * NH);       // (16, 64)
  attention_kernel<<<g2, 256, 0, stream>>>(qh, kh, vt, cx);
  dim3 g3(8192 / BM, DM / BN);        // (64, 8)
  out_proj_kernel<<<g3, 256, 0, stream>>>(cx, wo, bo, out);
}
